// TemplateEmbedder_17952963297600
// MI455X (gfx1250) — matrix operand order repaired and checked by reading, compiled, not run on hardware
//
#include <hip/hip_runtime.h>
#include <cmath>

// ---------------------------------------------------------------------------
// Types for CDNA5 WMMA (wave32, 16x16x32 bf16 -> f32)
// ---------------------------------------------------------------------------
typedef __attribute__((ext_vector_type(16))) __bf16 v16bf;
typedef __attribute__((ext_vector_type(8)))  float  v8f;

union Frag16 { v16bf v; uint4 q[2]; };
union Pack8  { uint4 u; __bf16 h[8]; };

__device__ __forceinline__ v8f wmma_bf16(const v16bf& a, const v16bf& b, const v8f& c) {
    return __builtin_amdgcn_wmma_f32_16x16x32_bf16(false, a, false, b, (short)0, c, false, false);
}

__device__ __forceinline__ float wave_sum(float v) {
    #pragma unroll
    for (int o = 16; o > 0; o >>= 1) v += __shfl_xor(v, o, 32);
    return v;
}
__device__ __forceinline__ float wave_max(float v) {
    #pragma unroll
    for (int o = 16; o > 0; o >>= 1) v = fmaxf(v, __shfl_xor(v, o, 32));
    return v;
}
__device__ __forceinline__ float sigmoidf_(float x) { return 1.f / (1.f + __expf(-x)); }

// ---------------------------------------------------------------------------
// Generic bf16 WMMA GEMM:  C[M,N] = A[M,K] * B^T  where B is stored [N][K].
//   256 threads (8 waves); block tile 128x64; wave tile 32x32 (2x2 WMMA)
//   DUAL: two B matrices / two accumulators (gated sigmoid & SwiGLU)
// Requirements: M % 128 == 0, K % 32 == 0. N guarded.
// ---------------------------------------------------------------------------
enum { EPI_F32 = 0, EPI_F32T, EPI_BF16, EPI_SIG_BF16, EPI_GATED_BF16,
       EPI_SWIGLU_BF16, EPI_ADD_F32, EPI_GATEADD_F32 };

__device__ __forceinline__ void load_b_tile(const __bf16* __restrict__ Bb, __bf16* Bl,
                                            int N, int K, int bn, int kt, int tid) {
    for (int c = tid; c < 64 * 4; c += 256) {
        int nrow = c >> 2, kc = (c & 3) * 8;
        uint4 val = {0u, 0u, 0u, 0u};
        if (bn + nrow < N)
            val = *(const uint4*)(Bb + (long)(bn + nrow) * K + kt + kc);
        *(uint4*)&Bl[nrow * 40 + kc] = val;
    }
}

template<int EPI, bool DUAL>
__global__ void __launch_bounds__(256)
gemm_bf16_kernel(const __bf16* __restrict__ A,
                 const __bf16* __restrict__ B1,
                 const __bf16* __restrict__ B2,
                 int M, int Nn, int K,
                 long strideA, long strideB, long strideC,
                 const float* __restrict__ bias,
                 const float* __restrict__ maskRow,
                 const __bf16* __restrict__ gate,
                 float* __restrict__ Cf,
                 __bf16* __restrict__ Cb)
{
    constexpr int BM = 128, BK = 32, LDA = 40, LDB = 40;
    __shared__ __bf16 Al[BM * LDA];
    __shared__ __bf16 Bl1[64 * LDB];
    __shared__ __bf16 Bl2[DUAL ? (64 * LDB) : 8];

    const int tid  = threadIdx.x;
    const int wid  = tid >> 5, lane = tid & 31;
    const int wm   = wid >> 1, wn = wid & 1;
    const int bm   = blockIdx.x * BM;
    const int bn   = blockIdx.y * 64;
    const int bz   = blockIdx.z;

    const __bf16* Ab  = A  + (long)bz * strideA;
    const __bf16* B1b = B1 + (long)bz * strideB;
    const __bf16* B2b = DUAL ? (B2 + (long)bz * strideB) : nullptr;

    v8f acc[2][2], acc2[2][2];
    #pragma unroll
    for (int a = 0; a < 2; ++a)
        #pragma unroll
        for (int b = 0; b < 2; ++b) { acc[a][b] = {}; acc2[a][b] = {}; }

    const int laneM = lane & 15;
    const int kb8   = (lane >> 4) * 8;
    const int klo   = (lane >> 4) * 16;

    for (int kt = 0; kt < K; kt += BK) {
        for (int c = tid; c < BM * BK / 8; c += 256) {
            int row = c >> 2, kc = (c & 3) * 8;
            *(uint4*)&Al[row * LDA + kc] =
                *(const uint4*)(Ab + (long)(bm + row) * K + kt + kc);
        }
        load_b_tile(B1b, Bl1, Nn, K, bn, kt, tid);
        if constexpr (DUAL) load_b_tile(B2b, Bl2, Nn, K, bn, kt, tid);
        __syncthreads();

        Frag16 af[2], bf1[2], bf2[2];
        #pragma unroll
        for (int a = 0; a < 2; ++a) {
            int r = wm * 32 + a * 16 + laneM;
            af[a].q[0] = *(const uint4*)&Al[r * LDA + klo];
            af[a].q[1] = *(const uint4*)&Al[r * LDA + klo + 8];
        }
        #pragma unroll
        for (int b = 0; b < 2; ++b) {
            int col = wn * 32 + b * 16 + laneM;
            bf1[b].q[0] = *(const uint4*)&Bl1[col * LDB + klo];
            bf1[b].q[1] = *(const uint4*)&Bl1[col * LDB + klo + 8];
            if constexpr (DUAL) {
                bf2[b].q[0] = *(const uint4*)&Bl2[col * LDB + klo];
                bf2[b].q[1] = *(const uint4*)&Bl2[col * LDB + klo + 8];
            }
        }
        #pragma unroll
        for (int a = 0; a < 2; ++a)
            #pragma unroll
            for (int b = 0; b < 2; ++b) {
                acc[a][b] = wmma_bf16(af[a].v, bf1[b].v, acc[a][b]);
                if constexpr (DUAL)
                    acc2[a][b] = wmma_bf16(af[a].v, bf2[b].v, acc2[a][b]);
            }
        __syncthreads();
    }

    // epilogue: C layout row = r + 8*(lane>>4), col = lane&15
    #pragma unroll
    for (int a = 0; a < 2; ++a) {
        #pragma unroll
        for (int b = 0; b < 2; ++b) {
            int col = bn + wn * 32 + b * 16 + laneM;
            if (col >= Nn) continue;
            const float bv = bias ? bias[col] : 0.f;
            const long rowBase = bm + wm * 32 + a * 16 + 8 * (lane >> 4);
            const long idx0 = (long)bz * strideC + rowBase * (long)Nn + col;

            if constexpr (EPI == EPI_F32) {
                #pragma unroll
                for (int r = 0; r < 8; ++r) Cf[idx0 + (long)r * Nn] = acc[a][b][r] + bv;
            } else if constexpr (EPI == EPI_F32T) {
                const long t0 = (long)col * M + rowBase;
                #pragma unroll
                for (int r = 0; r < 8; ++r) Cf[t0 + r] = acc[a][b][r] + bv;
            } else if constexpr (EPI == EPI_BF16) {
                #pragma unroll
                for (int r = 0; r < 8; ++r) Cb[idx0 + (long)r * Nn] = (__bf16)(acc[a][b][r] + bv);
            } else if constexpr (EPI == EPI_SIG_BF16) {
                #pragma unroll
                for (int r = 0; r < 8; ++r)
                    Cb[idx0 + (long)r * Nn] = (__bf16)sigmoidf_(acc[a][b][r] + bv);
            } else if constexpr (EPI == EPI_GATED_BF16) {
                float mr[8];
                #pragma unroll
                for (int r = 0; r < 8; ++r) mr[r] = maskRow ? maskRow[rowBase + r] : 1.f;
                #pragma unroll
                for (int r = 0; r < 8; ++r)
                    Cb[idx0 + (long)r * Nn] =
                        (__bf16)(sigmoidf_(acc[a][b][r]) * acc2[a][b][r] * mr[r]);
            } else if constexpr (EPI == EPI_SWIGLU_BF16) {
                #pragma unroll
                for (int r = 0; r < 8; ++r) {
                    float v1 = acc[a][b][r];
                    Cb[idx0 + (long)r * Nn] = (__bf16)(v1 * sigmoidf_(v1) * acc2[a][b][r]);
                }
            } else if constexpr (EPI == EPI_ADD_F32) {
                float prev[8];
                #pragma unroll
                for (int r = 0; r < 8; ++r) prev[r] = Cf[idx0 + (long)r * Nn];
                #pragma unroll
                for (int r = 0; r < 8; ++r) Cf[idx0 + (long)r * Nn] = prev[r] + acc[a][b][r] + bv;
            } else { // EPI_GATEADD_F32
                float prev[8], gv[8];
                #pragma unroll
                for (int r = 0; r < 8; ++r) prev[r] = Cf[idx0 + (long)r * Nn];
                #pragma unroll
                for (int r = 0; r < 8; ++r) gv[r] = (float)gate[idx0 + (long)r * Nn];
                #pragma unroll
                for (int r = 0; r < 8; ++r)
                    Cf[idx0 + (long)r * Nn] = prev[r] + gv[r] * acc[a][b][r];
            }
        }
    }
}

// ---------------------------------------------------------------------------
// Fused attention: workgroup = (problem, 32 query rows).
// Bias head-major [h][qrow][key]; bias+mask staged into an LDS fp32 tile.
// S-phase: dual WMMA issue, clause-batched LDS bias reads.
// O-phase: all 8 waves (key-split halves + LDS partial reduction).
// ---------------------------------------------------------------------------
__global__ void __launch_bounds__(256)
attn_kernel(const __bf16* __restrict__ q, const __bf16* __restrict__ k,
            const __bf16* __restrict__ v, const __bf16* __restrict__ gate,
            __bf16* __restrict__ o,
            const float* __restrict__ bias, const float* __restrict__ mask,
            int n, int d, int rowStride, int numH,
            long iStride, int hStride, long biasPlane, float scale)
{
    extern __shared__ __align__(16) char smem_raw[];
    float*  Sb = (float*)smem_raw;                         // 32*n f32 (bias+mask, then O partials)
    __bf16* Ql = (__bf16*)(smem_raw + (size_t)32 * n * 4); // 32*40
    __bf16* Kl = Ql + 32 * 40;                             // n*40
    __bf16* Vt = Kl + (long)n * 40;                        // 32*(n+8)  Vt[dd][kk]
    __bf16* Sl = Vt + 32L * (n + 8);                       // 32*(n+8)  S then P

    const int prob  = blockIdx.y;
    const int i_idx = prob / numH;
    const int h     = prob % numH;
    const long base = (long)i_idx * iStride + (long)h * hStride;
    const int  qt   = blockIdx.x * 32;
    const int  ldk  = n + 8;

    const int tid = threadIdx.x, wid = tid >> 5, lane = tid & 31, laneM = lane & 15;
    const int kb8 = (lane >> 4) * 8, klo = (lane >> 4) * 16;

    // ---- stage bias(+mask) tile ----
    {
        const float* bp = bias + (long)h * biasPlane + (long)qt * n;
        const int nch = n >> 2;
        for (int c = tid; c < 32 * nch; c += 256) {
            int r = c / nch, ch = (c % nch) * 4;
            float4 bvv = *(const float4*)(bp + (long)r * n + ch);
            if (mask) {
                float4 mv = *(const float4*)(mask + (long)i_idx * n + ch);
                bvv.x += (mv.x - 1.f) * 1e9f; bvv.y += (mv.y - 1.f) * 1e9f;
                bvv.z += (mv.z - 1.f) * 1e9f; bvv.w += (mv.w - 1.f) * 1e9f;
            }
            *(float4*)&Sb[r * n + ch] = bvv;
        }
    }
    // ---- stage Q ----
    for (int c = tid; c < 32 * 4; c += 256) {
        int r = c >> 2, d8 = (c & 3) * 8;
        uint4 val = {0u, 0u, 0u, 0u};
        if (d8 + 8 <= d) val = *(const uint4*)(q + base + (long)(qt + r) * rowStride + d8);
        else if (d8 < d) {
            Pack8 t; t.u = val;
            for (int e = 0; e < 8; ++e)
                t.h[e] = (d8 + e < d) ? q[base + (long)(qt + r) * rowStride + d8 + e] : (__bf16)0.f;
            val = t.u;
        }
        *(uint4*)&Ql[r * 40 + d8] = val;
    }
    // ---- stage K ----
    for (int c = tid; c < n * 4; c += 256) {
        int kk = c >> 2, d8 = (c & 3) * 8;
        uint4 val = {0u, 0u, 0u, 0u};
        if (d8 + 8 <= d) val = *(const uint4*)(k + base + (long)kk * rowStride + d8);
        else if (d8 < d) {
            Pack8 t; t.u = val;
            for (int e = 0; e < 8; ++e)
                t.h[e] = (d8 + e < d) ? k[base + (long)kk * rowStride + d8 + e] : (__bf16)0.f;
            val = t.u;
        }
        *(uint4*)&Kl[kk * 40 + d8] = val;
    }
    // ---- stage V transposed ----
    for (int c = tid; c < n * 4; c += 256) {
        int kk = c >> 2, d8 = (c & 3) * 8;
        Pack8 t; t.u = make_uint4(0u, 0u, 0u, 0u);
        if (d8 + 8 <= d) t.u = *(const uint4*)(v + base + (long)kk * rowStride + d8);
        else if (d8 < d)
            for (int e = 0; e < 8; ++e)
                t.h[e] = (d8 + e < d) ? v[base + (long)kk * rowStride + d8 + e] : (__bf16)0.f;
        #pragma unroll
        for (int e = 0; e < 8; ++e) Vt[(d8 + e) * ldk + kk] = t.h[e];
    }
    __syncthreads();

    // ---- S = scale*Q*K^T + bias tile ----
    Frag16 af[2];
    #pragma unroll
    for (int a = 0; a < 2; ++a) {
        int r = a * 16 + laneM;
        af[a].q[0] = *(const uint4*)&Ql[r * 40 + klo];
        af[a].q[1] = *(const uint4*)&Ql[r * 40 + klo + 8];
    }
    const int nt = n >> 4;
    for (int ct = wid; ct < nt; ct += 8) {
        Frag16 bfr;
        int col = ct * 16 + laneM;
        bfr.q[0] = *(const uint4*)&Kl[col * 40 + klo];
        bfr.q[1] = *(const uint4*)&Kl[col * 40 + klo + 8];
        v8f z0 = {}, z1 = {};
        v8f sacc0 = wmma_bf16(af[0].v, bfr.v, z0);
        v8f sacc1 = wmma_bf16(af[1].v, bfr.v, z1);
        const int kk = ct * 16 + laneM;
        const int r0 = 8 * (lane >> 4);
        float bv0[8], bv1[8];
        #pragma unroll
        for (int r = 0; r < 8; ++r) bv0[r] = Sb[(r0 + r) * n + kk];
        #pragma unroll
        for (int r = 0; r < 8; ++r) bv1[r] = Sb[(16 + r0 + r) * n + kk];
        #pragma unroll
        for (int r = 0; r < 8; ++r)
            Sl[(r0 + r) * ldk + kk] = (__bf16)(sacc0[r] * scale + bv0[r]);
        #pragma unroll
        for (int r = 0; r < 8; ++r)
            Sl[(16 + r0 + r) * ldk + kk] = (__bf16)(sacc1[r] * scale + bv1[r]);
    }
    __syncthreads();

    // ---- softmax over keys (each wave: 4 rows) ----
    {
        float vals[12];
        const int cnt = n / 32;
        for (int rr = 0; rr < 4; ++rr) {
            int rloc = wid * 4 + rr;
            for (int t = 0; t < cnt; ++t)
                vals[t] = (float)Sl[rloc * ldk + t * 32 + lane];
            float mx = -1e30f;
            for (int t = 0; t < cnt; ++t) mx = fmaxf(mx, vals[t]);
            mx = wave_max(mx);
            float sum = 0.f;
            for (int t = 0; t < cnt; ++t) { vals[t] = __expf(vals[t] - mx); sum += vals[t]; }
            sum = wave_sum(sum);
            float inv = 1.f / sum;
            for (int t = 0; t < cnt; ++t)
                Sl[rloc * ldk + t * 32 + lane] = (__bf16)(vals[t] * inv);
        }
    }
    __syncthreads();

    // ---- O = P V: 8 waves, keys split into halves; partial reduce via LDS ----
    {
        const int tileId = wid & 3;
        const int a = tileId >> 1, bcol = tileId & 1;
        const int half = wid >> 2;
        const int nh = n >> 1;
        v8f acc = {};
        for (int kt = half * nh; kt < half * nh + nh; kt += 32) {
            Frag16 pa, vb;
            int r  = a * 16 + laneM;
            pa.q[0] = *(const uint4*)&Sl[r * ldk + kt + klo];
            pa.q[1] = *(const uint4*)&Sl[r * ldk + kt + klo + 8];
            int dd = bcol * 16 + laneM;
            vb.q[0] = *(const uint4*)&Vt[dd * ldk + kt + klo];
            vb.q[1] = *(const uint4*)&Vt[dd * ldk + kt + klo + 8];
            acc = wmma_bf16(pa.v, vb.v, acc);
        }
        if (wid >= 4) {
            #pragma unroll
            for (int r = 0; r < 8; ++r)
                Sb[tileId * 256 + (r + 8 * (lane >> 4)) * 16 + laneM] = acc[r];
        }
        __syncthreads();
        if (wid < 4) {
            float part[8];
            #pragma unroll
            for (int r = 0; r < 8; ++r)
                part[r] = Sb[tileId * 256 + (r + 8 * (lane >> 4)) * 16 + laneM];
            int dd = bcol * 16 + laneM;
            if (dd < d) {
                const long i0 = base + (long)(qt + a * 16 + 8 * (lane >> 4)) * rowStride + dd;
                float gv[8];
                if (gate) {
                    #pragma unroll
                    for (int r = 0; r < 8; ++r) gv[r] = (float)gate[i0 + (long)r * rowStride];
                }
                #pragma unroll
                for (int r = 0; r < 8; ++r) {
                    float val = acc[r] + part[r];
                    if (gate) val *= gv[r];
                    o[i0 + (long)r * rowStride] = (__bf16)val;
                }
            }
        }
    }
}

// ---------------------------------------------------------------------------
// LayerNorm over last dim (C in {128,384}); one wave per row; bf16 out.
// ---------------------------------------------------------------------------
template<typename TIN>
__global__ void __launch_bounds__(256)
ln_kernel(const TIN* __restrict__ x, __bf16* __restrict__ out,
          const float* __restrict__ g, const float* __restrict__ b,
          int C, long rows)
{
    const int wid = threadIdx.x >> 5, lane = threadIdx.x & 31;
    const long row = (long)blockIdx.x * 8 + wid;
    if (row >= rows) return;
    const TIN* xr = x + row * C;
    const int per = C / 32;
    float vals[12];
    for (int t = 0; t < per; ++t) vals[t] = (float)xr[t * 32 + lane];
    float m = 0.f;
    for (int t = 0; t < per; ++t) m += vals[t];
    m = wave_sum(m) / (float)C;
    float var = 0.f;
    for (int t = 0; t < per; ++t) { float dv = vals[t] - m; var += dv * dv; }
    var = wave_sum(var) / (float)C;
    float inv = rsqrtf(var + 1e-5f);
    for (int t = 0; t < per; ++t) {
        int c = t * 32 + lane;
        out[row * C + c] = (__bf16)((vals[t] - m) * inv * g[c] + b[c]);
    }
}

// ---------------------------------------------------------------------------
// Weight convert + transpose: src fp32 [K][N]  ->  dst bf16 [N][K]
// ---------------------------------------------------------------------------
__global__ void cvtT_kernel(const float* __restrict__ src, __bf16* __restrict__ dst,
                            int K, int N)
{
    const long total = (long)K * N;
    for (long i = (long)blockIdx.x * 256 + threadIdx.x; i < total; i += (long)gridDim.x * 256) {
        int nn = (int)(i / K), kk = (int)(i % K);
        dst[i] = (__bf16)src[(long)kk * N + nn];
    }
}

// ---------------------------------------------------------------------------
// Tiled transposes (all global accesses 16B vectors)
// ---------------------------------------------------------------------------
__global__ void __launch_bounds__(256) to_cmajor_kernel(const __bf16* __restrict__ src,
                                                        __bf16* __restrict__ dst)
{
    constexpr long R = 147456;
    __shared__ __bf16 T[64 * 136];
    const long rbase = (long)blockIdx.x * 64;
    const int tid = threadIdx.x;
    for (int i = tid; i < 1024; i += 256) {
        int row = i >> 4, cc = (i & 15) * 8;
        *(uint4*)&T[row * 136 + cc] = *(const uint4*)(src + (rbase + row) * 128 + cc);
    }
    __syncthreads();
    for (int i = tid; i < 1024; i += 256) {
        int c = i >> 3, rc8 = (i & 7) * 8;
        Pack8 t;
        #pragma unroll
        for (int e = 0; e < 8; ++e) t.h[e] = T[(rc8 + e) * 136 + c];
        *(uint4*)(dst + (long)c * R + rbase + rc8) = t.u;
    }
}

__global__ void __launch_bounds__(256) from_cmajor_kernel(const __bf16* __restrict__ src,
                                                          __bf16* __restrict__ dst)
{
    constexpr long R = 147456;
    __shared__ __bf16 T[64 * 136];
    const long rbase = (long)blockIdx.x * 64;
    const int tid = threadIdx.x;
    for (int i = tid; i < 1024; i += 256) {
        int c = i >> 3, rc8 = (i & 7) * 8;
        Pack8 t;
        t.u = *(const uint4*)(src + (long)c * R + rbase + rc8);
        #pragma unroll
        for (int e = 0; e < 8; ++e) T[(rc8 + e) * 136 + c] = t.h[e];
    }
    __syncthreads();
    for (int i = tid; i < 1024; i += 256) {
        int row = i >> 4, cc = (i & 15) * 8;
        *(uint4*)(dst + (rbase + row) * 128 + cc) = *(uint4*)&T[row * 136 + cc];
    }
}

// per-channel 384x384 transpose: dst[c][x][y] = src[c][y][x]
__global__ void __launch_bounds__(256) ct384_kernel(const __bf16* __restrict__ src,
                                                    __bf16* __restrict__ dst)
{
    constexpr long R = 147456;
    __shared__ __bf16 T[64 * 72];
    const int xb = blockIdx.x * 64, yb = blockIdx.y * 64, c = blockIdx.z;
    const int tid = threadIdx.x;
    for (int i = tid; i < 512; i += 256) {
        int r = i >> 3, cc = (i & 7) * 8;
        *(uint4*)&T[r * 72 + cc] =
            *(const uint4*)(src + (long)c * R + (long)(yb + r) * 384 + xb + cc);
    }
    __syncthreads();
    for (int i = tid; i < 512; i += 256) {
        int r2 = i >> 3, cc2 = (i & 7) * 8;
        Pack8 t;
        #pragma unroll
        for (int e = 0; e < 8; ++e) t.h[e] = T[(cc2 + e) * 72 + r2];
        *(uint4*)(dst + (long)c * R + (long)(xb + r2) * 384 + yb + cc2) = t.u;
    }
}

// swap (i,j) of [384][384][128] fp32
__global__ void ztrans_kernel(const float* __restrict__ src, float* __restrict__ dst)
{
    long p = blockIdx.x;
    int j = (int)(p % 384), i = (int)(p / 384);
    dst[((long)j * 384 + i) * 128 + threadIdx.x] = src[p * 128 + threadIdx.x];
}

// ---------------------------------------------------------------------------
// Host orchestration
// ---------------------------------------------------------------------------
extern "C" void kernel_launch(void* const* d_in, const int* in_sizes, int n_in,
                              void* d_out, int out_size, void* d_ws, size_t ws_size,
                              hipStream_t stream)
{
    const int  N  = 384, CZ = 128, CS = 384;
    const long R  = (long)N * N;
    const long RC = R * CZ;

    const float* s_in = (const float*)d_in[0];
    const float* z_in = (const float*)d_in[1];
    const float* mask = (const float*)d_in[2];
    auto P = [&](int i) { return (const float*)d_in[i]; };

    float* s = (float*)d_out;
    float* z = s + R;
    hipMemcpyAsync(s, s_in, R * sizeof(float),  hipMemcpyDeviceToDevice, stream);
    hipMemcpyAsync(z, z_in, RC * sizeof(float), hipMemcpyDeviceToDevice, stream);

    char* wp = (char*)d_ws;
    auto alloc = [&](size_t bytes) -> void* {
        void* p = wp; wp += (bytes + 255) & ~(size_t)255; return p;
    };
    auto cvtT = [&](int idx, int K, int Nn) -> __bf16* {
        __bf16* dst = (__bf16*)alloc((size_t)K * Nn * 2);
        cvtT_kernel<<<256, 256, 0, stream>>>(P(idx), dst, K, Nn);
        return dst;
    };

    __bf16* pool = (__bf16*)alloc((size_t)RC * 2 * 6);
    __bf16* S0 = pool;          __bf16* S1 = pool + RC;
    __bf16* S2 = pool + 2 * RC; __bf16* S3 = pool + 3 * RC;
    __bf16* S4 = pool + 4 * RC; __bf16* S5 = pool + 5 * RC;
    float*  zt     = (float*)alloc((size_t)RC * 4);
    float*  triBuf = (float*)alloc((size_t)R * 4 * 4);
    float*  zbBuf  = (float*)alloc((size_t)R * 16 * 4);
    __bf16* ab_al = (__bf16*)alloc((size_t)R * 2);
    __bf16* ab_q  = (__bf16*)alloc((size_t)R * 2);
    __bf16* ab_k  = (__bf16*)alloc((size_t)R * 2);
    __bf16* ab_v  = (__bf16*)alloc((size_t)R * 2);
    __bf16* ab_g  = (__bf16*)alloc((size_t)R * 2);
    __bf16* ab_o  = (__bf16*)alloc((size_t)R * 2);

    auto gemm = [&](int epi, bool dual,
                    const __bf16* A, const __bf16* B1, const __bf16* B2,
                    long M, int Nn, int K, long sa, long sb, long sc, int batch,
                    const float* bias, const float* maskRow, const __bf16* gate,
                    float* Cf, __bf16* Cb) {
        dim3 grid((unsigned)(M / 128), (unsigned)((Nn + 63) / 64), (unsigned)batch);
        dim3 block(256);
        if (dual) {
            if (epi == EPI_GATED_BF16)
                gemm_bf16_kernel<EPI_GATED_BF16, true><<<grid, block, 0, stream>>>(
                    A, B1, B2, (int)M, Nn, K, sa, sb, sc, bias, maskRow, gate, Cf, Cb);
            else
                gemm_bf16_kernel<EPI_SWIGLU_BF16, true><<<grid, block, 0, stream>>>(
                    A, B1, B2, (int)M, Nn, K, sa, sb, sc, bias, maskRow, gate, Cf, Cb);
            return;
        }
        switch (epi) {
        case EPI_F32:
            gemm_bf16_kernel<EPI_F32, false><<<grid, block, 0, stream>>>(
                A, B1, B2, (int)M, Nn, K, sa, sb, sc, bias, maskRow, gate, Cf, Cb); break;
        case EPI_F32T:
            gemm_bf16_kernel<EPI_F32T, false><<<grid, block, 0, stream>>>(
                A, B1, B2, (int)M, Nn, K, sa, sb, sc, bias, maskRow, gate, Cf, Cb); break;
        case EPI_BF16:
            gemm_bf16_kernel<EPI_BF16, false><<<grid, block, 0, stream>>>(
                A, B1, B2, (int)M, Nn, K, sa, sb, sc, bias, maskRow, gate, Cf, Cb); break;
        case EPI_SIG_BF16:
            gemm_bf16_kernel<EPI_SIG_BF16, false><<<grid, block, 0, stream>>>(
                A, B1, B2, (int)M, Nn, K, sa, sb, sc, bias, maskRow, gate, Cf, Cb); break;
        case EPI_ADD_F32:
            gemm_bf16_kernel<EPI_ADD_F32, false><<<grid, block, 0, stream>>>(
                A, B1, B2, (int)M, Nn, K, sa, sb, sc, bias, maskRow, gate, Cf, Cb); break;
        default:
            gemm_bf16_kernel<EPI_GATEADD_F32, false><<<grid, block, 0, stream>>>(
                A, B1, B2, (int)M, Nn, K, sa, sb, sc, bias, maskRow, gate, Cf, Cb); break;
        }
    };

    auto ln_f32 = [&](const float* x, __bf16* out, const float* g, const float* b,
                      int C, long rows) {
        ln_kernel<float><<<dim3((unsigned)(rows / 8)), 256, 0, stream>>>(x, out, g, b, C, rows);
    };
    auto ln_bf16 = [&](const __bf16* x, __bf16* out, const float* g, const float* b,
                       int C, long rows) {
        ln_kernel<__bf16><<<dim3((unsigned)(rows / 8)), 256, 0, stream>>>(x, out, g, b, C, rows);
    };

    const size_t attn_shmem =
        (size_t)32 * 384 * 4 + (size_t)(32 * 40 + 384 * 40 + 32 * 392 + 32 * 392) * 2;

    auto tri_mul = [&](float* zb_, int bi, bool outgoing) {
        __bf16* wag = cvtT(bi + 2, CZ, CZ);
        __bf16* wap = cvtT(bi + 3, CZ, CZ);
        __bf16* wbg = cvtT(bi + 4, CZ, CZ);
        __bf16* wbp = cvtT(bi + 5, CZ, CZ);
        __bf16* wg  = cvtT(bi + 6, CZ, CZ);
        __bf16* wo  = cvtT(bi + 9, CZ, CZ);
        ln_f32(zb_, S0, P(bi + 0), P(bi + 1), CZ, R);
        gemm(EPI_GATED_BF16, true, S0, wag, wap, R, CZ, CZ, 0, 0, 0, 1,
             nullptr, mask, nullptr, nullptr, S1);
        gemm(EPI_GATED_BF16, true, S0, wbg, wbp, R, CZ, CZ, 0, 0, 0, 1,
             nullptr, mask, nullptr, nullptr, S2);
        gemm(EPI_SIG_BF16, false, S0, wg, nullptr, R, CZ, CZ, 0, 0, 0, 1,
             nullptr, nullptr, nullptr, nullptr, S3);
        const unsigned TB = (unsigned)(R / 64);
        if (outgoing) {
            to_cmajor_kernel<<<TB, 256, 0, stream>>>(S1, S4);
            to_cmajor_kernel<<<TB, 256, 0, stream>>>(S2, S0);
        } else {
            to_cmajor_kernel<<<TB, 256, 0, stream>>>(S1, S5);
            ct384_kernel<<<dim3(6, 6, 128), 256, 0, stream>>>(S5, S4);
            to_cmajor_kernel<<<TB, 256, 0, stream>>>(S2, S5);
            ct384_kernel<<<dim3(6, 6, 128), 256, 0, stream>>>(S5, S0);
        }
        gemm(EPI_BF16, false, S4, S0, nullptr, N, N, N, R, R, R, CZ,
             nullptr, nullptr, nullptr, nullptr, S1);
        from_cmajor_kernel<<<TB, 256, 0, stream>>>(S1, S2);
        ln_bf16(S2, S1, P(bi + 7), P(bi + 8), CZ, R);
        gemm(EPI_GATEADD_F32, false, S1, wo, nullptr, R, CZ, CZ, 0, 0, 0, 1,
             nullptr, nullptr, S3, zb_, nullptr);
    };

    auto tri_att = [&](float* zb_, int bi, const float* maskp) {
        __bf16* wq = cvtT(bi + 2, CZ, 128);
        __bf16* wk = cvtT(bi + 3, CZ, 128);
        __bf16* wv = cvtT(bi + 4, CZ, 128);
        __bf16* wb = cvtT(bi + 5, CZ, 4);
        __bf16* wg = cvtT(bi + 6, CZ, 128);
        __bf16* wo = cvtT(bi + 7, 128, CZ);
        const float* bo = P(bi + 8);
        ln_f32(zb_, S0, P(bi + 0), P(bi + 1), CZ, R);
        gemm(EPI_BF16, false, S0, wq, nullptr, R, 128, CZ, 0, 0, 0, 1,
             nullptr, nullptr, nullptr, nullptr, S1);
        gemm(EPI_BF16, false, S0, wk, nullptr, R, 128, CZ, 0, 0, 0, 1,
             nullptr, nullptr, nullptr, nullptr, S2);
        gemm(EPI_BF16, false, S0, wv, nullptr, R, 128, CZ, 0, 0, 0, 1,
             nullptr, nullptr, nullptr, nullptr, S3);
        gemm(EPI_SIG_BF16, false, S0, wg, nullptr, R, 128, CZ, 0, 0, 0, 1,
             nullptr, nullptr, nullptr, nullptr, S4);
        gemm(EPI_F32T, false, S0, wb, nullptr, R, 4, CZ, 0, 0, 0, 1,
             nullptr, nullptr, nullptr, triBuf, nullptr);
        attn_kernel<<<dim3(12, 384 * 4), 256, attn_shmem, stream>>>(
            S1, S2, S3, S4, S0, triBuf, maskp,
            384, 32, 128, 4, (long)384 * 128, 32, R, 1.f / sqrtf(32.f));
        gemm(EPI_ADD_F32, false, S0, wo, nullptr, R, CZ, 128, 0, 0, 0, 1,
             bo, nullptr, nullptr, zb_, nullptr);
    };

    auto transition = [&](float* xptr, int bi, int C, long rows) {
        __bf16* w1 = cvtT(bi + 2, C, 4 * C);
        __bf16* w2 = cvtT(bi + 3, C, 4 * C);
        __bf16* w3 = cvtT(bi + 4, 4 * C, C);
        ln_f32(xptr, S0, P(bi + 0), P(bi + 1), C, rows);
        gemm(EPI_SWIGLU_BF16, true, S0, w1, w2, rows, 4 * C, C, 0, 0, 0, 1,
             nullptr, nullptr, nullptr, nullptr, S1);
        gemm(EPI_ADD_F32, false, S1, w3, nullptr, rows, C, 4 * C, 0, 0, 0, 1,
             nullptr, nullptr, nullptr, xptr, nullptr);
    };

    // ================= Pairformer block =================
    tri_mul(z, 3, true);
    tri_mul(z, 13, false);
    tri_att(z, 23, mask);
    ztrans_kernel<<<(unsigned)R, 128, 0, stream>>>(z, zt);
    tri_att(zt, 32, mask);
    ztrans_kernel<<<(unsigned)R, 128, 0, stream>>>(zt, z);
    transition(z, 41, CZ, R);

    {
        __bf16* wq = cvtT(48, CS, CS); const float* bq = P(49);
        __bf16* wk = cvtT(50, CS, CS);
        __bf16* wv = cvtT(51, CS, CS);
        __bf16* wb = cvtT(54, CZ, 16);
        __bf16* wg = cvtT(55, CS, CS);
        __bf16* wo = cvtT(56, CS, CS);
        ln_f32(s, ab_al, P(46), P(47), CS, N);
        gemm(EPI_BF16, false, ab_al, wq, nullptr, N, CS, CS, 0, 0, 0, 1,
             bq, nullptr, nullptr, nullptr, ab_q);
        gemm(EPI_BF16, false, ab_al, wk, nullptr, N, CS, CS, 0, 0, 0, 1,
             nullptr, nullptr, nullptr, nullptr, ab_k);
        gemm(EPI_BF16, false, ab_al, wv, nullptr, N, CS, CS, 0, 0, 0, 1,
             nullptr, nullptr, nullptr, nullptr, ab_v);
        gemm(EPI_SIG_BF16, false, ab_al, wg, nullptr, N, CS, CS, 0, 0, 0, 1,
             nullptr, nullptr, nullptr, nullptr, ab_g);
        ln_f32(z, S0, P(52), P(53), CZ, R);
        gemm(EPI_F32T, false, S0, wb, nullptr, R, 16, CZ, 0, 0, 0, 1,
             nullptr, nullptr, nullptr, zbBuf, nullptr);
        attn_kernel<<<dim3(12, 16), 256, attn_shmem, stream>>>(
            ab_q, ab_k, ab_v, ab_g, ab_o, zbBuf, nullptr,
            384, 24, 384, 16, 0L, 24, R, 1.f / sqrtf(24.f));
        gemm(EPI_ADD_F32, false, ab_o, wo, nullptr, N, CS, CS, 0, 0, 0, 1,
             nullptr, nullptr, nullptr, s, nullptr);
    }

    transition(s, 57, CS, N);
    (void)in_sizes; (void)n_in; (void)out_size; (void)ws_size;
}
